// CoherenceAttention_77695958384924
// MI455X (gfx1250) — compile-verified, matrix operand order checked by reading, not run on hardware
//
#include <hip/hip_runtime.h>
#include <hip/hip_bf16.h>
#include <stdint.h>

// ---------------------------------------------------------------------------
// CoherenceAttention for MI455X (gfx1250): LN -> QKV (WMMA f16) ->
// flash-attention (WMMA f16, online softmax) -> out-proj + residual (WMMA).
// Round 3: fix async-to-lds builtin signature (param0 probe-confirmed:
// addrspace(1) pointer to int __attribute__((vector_size(16)))).
// ---------------------------------------------------------------------------

#define DIM   768
#define BATCH 4
#define SEQ   2048
#define ROWS  (BATCH * SEQ)          // 8192
#define LN_EPS 1e-5f
// softmax done in base-2: logits pre-scaled by (1/sqrt(D)) * log2(e)
#define SM_SCALE_LOG2E (0.03608439182435161f * 1.4426950408889634f)

typedef __attribute__((ext_vector_type(16))) _Float16 v16h;
typedef __attribute__((ext_vector_type(4)))  _Float16 v4h;
typedef __attribute__((ext_vector_type(8)))  float    v8f;

// ---------------- gfx1250 async-to-LDS (probe-confirmed builtin) -----------
#if __has_builtin(__builtin_amdgcn_global_load_async_to_lds_b128) && \
    __has_builtin(__builtin_amdgcn_s_wait_asynccnt)
#define ASYNC_LDS 1
typedef int v4i32 __attribute__((vector_size(16)));
typedef __attribute__((address_space(1))) v4i32 gv4i_t;   // global (AS1)
typedef __attribute__((address_space(3))) v4i32 lv4i_t;   // LDS (AS3)
#else
#define ASYNC_LDS 0
#endif

// 16-byte global -> LDS copy (async if available, else b128 load+store).
__device__ __forceinline__ void copy16_g2l(void* lds_dst, const void* gsrc) {
#if ASYNC_LDS
  __builtin_amdgcn_global_load_async_to_lds_b128(
      (gv4i_t*)(uintptr_t)gsrc, (lv4i_t*)(uintptr_t)lds_dst, 0, 0);
#else
  *(uint4*)lds_dst = *(const uint4*)gsrc;
#endif
}
__device__ __forceinline__ void async_join() {
#if ASYNC_LDS
  __builtin_amdgcn_s_wait_asynccnt(0);
#endif
}

__device__ __forceinline__ float fast_exp2(float x) {
#if __has_builtin(__builtin_amdgcn_exp2f)
  return __builtin_amdgcn_exp2f(x);
#else
  return exp2f(x);
#endif
}

__device__ __forceinline__ v8f vzero8() {
  v8f z = {0.f, 0.f, 0.f, 0.f, 0.f, 0.f, 0.f, 0.f};
  return z;
}

// WMMA wrapper: D = A(16x32 f16) * B(32x16 f16) + C(16x16 f32)
__device__ __forceinline__ v8f wmma_f16(v16h a, v16h b, v8f c) {
  return __builtin_amdgcn_wmma_f32_16x16x32_f16(
      false, a, false, b, (short)0, c, false, false);
}

// Load a 16x32 f16 A-fragment (or B-fragment stored [n][k]-major) from LDS.
// Per CDNA5 ISA 7.12.2: lane L holds row m=L&15; VGPR e holds
//   k = 16*(e>=4) + 8*(L>>4) + 2*(e&3) + {0,1}
__device__ __forceinline__ v16h load_frag_row(const _Float16* base, int stride,
                                              int lane) {
  const int m  = lane & 15;
  const int hi = lane >> 4;
  v16h a;
#pragma unroll
  for (int e = 0; e < 8; ++e) {
    const int k = ((e & 4) << 2) + (hi << 3) + ((e & 3) << 1);
    a[2 * e]     = base[m * stride + k];
    a[2 * e + 1] = base[m * stride + k + 1];
  }
  return a;
}

// Load a 32x16 f16 B-fragment from an LDS tile stored [k][n]-major.
__device__ __forceinline__ v16h load_frag_b_kn(const _Float16* base, int stride,
                                               int lane) {
  const int n  = lane & 15;
  const int hi = lane >> 4;
  v16h b;
#pragma unroll
  for (int e = 0; e < 8; ++e) {
    const int k = ((e & 4) << 2) + (hi << 3) + ((e & 3) << 1);
    b[2 * e]     = base[k * stride + n];
    b[2 * e + 1] = base[(k + 1) * stride + n];
  }
  return b;
}

// Reductions within each 16-lane half of a wave32.
__device__ __forceinline__ float half_reduce_max(float v) {
  v = fmaxf(v, __shfl_xor(v, 1, 32));
  v = fmaxf(v, __shfl_xor(v, 2, 32));
  v = fmaxf(v, __shfl_xor(v, 4, 32));
  v = fmaxf(v, __shfl_xor(v, 8, 32));
  return v;
}
__device__ __forceinline__ float half_reduce_sum(float v) {
  v += __shfl_xor(v, 1, 32);
  v += __shfl_xor(v, 2, 32);
  v += __shfl_xor(v, 4, 32);
  v += __shfl_xor(v, 8, 32);
  return v;
}

// ---------------------------------------------------------------------------
// Kernel 1: LayerNorm + cast to f16.  One wave32 per row, 8 rows per block.
// ---------------------------------------------------------------------------
__global__ void __launch_bounds__(256)
ln_cast_kernel(const float* __restrict__ x, const float* __restrict__ g,
               const float* __restrict__ b, _Float16* __restrict__ h) {
  const int row  = blockIdx.x * 8 + (threadIdx.x >> 5);
  const int lane = threadIdx.x & 31;
  const float* xr = x + (size_t)row * DIM;

  float vals[24];
  float s = 0.f;
#pragma unroll
  for (int i = 0; i < 24; ++i) {
    vals[i] = xr[lane + 32 * i];
    s += vals[i];
  }
#pragma unroll
  for (int off = 16; off > 0; off >>= 1) s += __shfl_xor(s, off, 32);
  const float mu = s * (1.0f / DIM);

  float vv = 0.f;
#pragma unroll
  for (int i = 0; i < 24; ++i) {
    const float d = vals[i] - mu;
    vv += d * d;
  }
#pragma unroll
  for (int off = 16; off > 0; off >>= 1) vv += __shfl_xor(vv, off, 32);
  const float rstd = rsqrtf(vv * (1.0f / DIM) + LN_EPS);

  _Float16* hr = h + (size_t)row * DIM;
#pragma unroll
  for (int i = 0; i < 24; ++i) {
    const int c = lane + 32 * i;
    hr[c] = (_Float16)((vals[i] - mu) * rstd * g[c] + b[c]);
  }
}

// ---------------------------------------------------------------------------
// Kernel 2: Y(8192x768 f16) = A(8192x768 f16) @ W^T(768x768 f32).
// Block tile 64x256, 8 waves as 2(M)x4(N), wave tile 32x64 (2x4 WMMA tiles).
// Double-buffered LDS staging, b128 copies, prefetch.
// ---------------------------------------------------------------------------
__device__ __forceinline__ void gemm_stage(_Float16 (*As)[40], _Float16 (*Bs)[40],
                                           const _Float16* __restrict__ A,
                                           const float* __restrict__ W,
                                           int m0, int n0, int kb, int tid) {
  // A: 64x32 halves, 4 threads/row, one b128 each.
  {
    const int r = tid >> 2, c = (tid & 3) * 8;
    const _Float16* src = A + (size_t)(m0 + r) * DIM + kb + c;
    copy16_g2l(&As[r][c], src);
    __builtin_prefetch(src + 32, 0, 1);  // next K-step
  }
  // W: 256 rows x 32 f32 -> f16, 1 thread/row.
  {
    const int r = tid;
    const float4* wsrc = (const float4*)(W + (size_t)(n0 + r) * DIM + kb);
    __builtin_prefetch(wsrc + 8, 0, 1);  // next K-step
#pragma unroll
    for (int t = 0; t < 8; ++t) {
      const float4 w = wsrc[t];
      v4h hv = {(_Float16)w.x, (_Float16)w.y, (_Float16)w.z, (_Float16)w.w};
      *(v4h*)&Bs[r][4 * t] = hv;
    }
  }
}

__global__ void __launch_bounds__(256)
gemm_wt_f16_kernel(const _Float16* __restrict__ A, const float* __restrict__ W,
                   _Float16* __restrict__ Y) {
  __shared__ _Float16 As[2][64][40];    // [m][k], +8 pad
  __shared__ _Float16 Bs[2][256][40];   // [n][k], +8 pad

  const int tid  = threadIdx.x;
  const int lane = tid & 31;
  const int wid  = tid >> 5;
  const int wm   = wid & 1;
  const int wn   = wid >> 1;
  const int m0   = blockIdx.x * 64;
  const int n0   = blockIdx.y * 256;

  v8f acc[2][4];
#pragma unroll
  for (int i = 0; i < 2; ++i)
#pragma unroll
    for (int j = 0; j < 4; ++j) acc[i][j] = vzero8();

  gemm_stage(As[0], Bs[0], A, W, m0, n0, 0, tid);
  async_join();
  __syncthreads();

#pragma unroll 2
  for (int it = 0; it < DIM / 32; ++it) {
    const int cur = it & 1;
    if (it + 1 < DIM / 32)
      gemm_stage(As[cur ^ 1], Bs[cur ^ 1], A, W, m0, n0, (it + 1) * 32, tid);

    v16h af[2], bf[4];
#pragma unroll
    for (int i = 0; i < 2; ++i)
      af[i] = load_frag_row(&As[cur][32 * wm + 16 * i][0], 40, lane);
#pragma unroll
    for (int j = 0; j < 4; ++j)
      bf[j] = load_frag_row(&Bs[cur][64 * wn + 16 * j][0], 40, lane);
#pragma unroll
    for (int i = 0; i < 2; ++i)
#pragma unroll
      for (int j = 0; j < 4; ++j) acc[i][j] = wmma_f16(af[i], bf[j], acc[i][j]);

    async_join();
    __syncthreads();
  }

  const int hi = lane >> 4, nl = lane & 15;
#pragma unroll
  for (int i = 0; i < 2; ++i)
#pragma unroll
    for (int j = 0; j < 4; ++j)
#pragma unroll
      for (int jj = 0; jj < 8; ++jj) {
        const int row = m0 + 32 * wm + 16 * i + jj + 8 * hi;
        const int col = n0 + 64 * wn + 16 * j + nl;
        Y[(size_t)row * DIM + col] = (_Float16)acc[i][j][jj];
      }
}

// ---------------------------------------------------------------------------
// Kernel 3: flash attention.  Workgroup = (b, 32 query rows), 8 waves.
// Scores: waves 2(M)x4(N) over 32x128 key block; K double-buffered 128x64.
// PV: waves partition 128-col V chunks (16 cols each), O persistent in VGPRs.
// ---------------------------------------------------------------------------
#define QS_STR 776   // 768 + 8
#define KS_STR 72    // 64 + 8
#define PS_STR 136   // 128 + 8
#define VS_STR 136
#define Q_HALVES (32 * QS_STR)
#define K_HALVES (128 * KS_STR)
#define P_HALVES (32 * PS_STR)
#define V_HALVES (128 * VS_STR)
#define ATTN_SMEM_BYTES \
  ((Q_HALVES + 2 * K_HALVES + P_HALVES + V_HALVES) * 2 + 352 * 4)

__device__ __forceinline__ void stage_k(_Float16* Ks,
                                        const _Float16* __restrict__ Kg,
                                        size_t rowbase, int kb, int dch,
                                        int tid) {
  const int kr  = tid >> 1;
  const int kc0 = (tid & 1) * 32;
  const _Float16* src = Kg + (rowbase + kb * 128 + kr) * DIM + dch * 64 + kc0;
  __builtin_prefetch(src + 64, 0, 1);  // next D-chunk
#pragma unroll
  for (int t = 0; t < 4; ++t)
    copy16_g2l(Ks + kr * KS_STR + kc0 + 8 * t, src + 8 * t);
}

__global__ void __launch_bounds__(256)
attn_kernel(const _Float16* __restrict__ Qg, const _Float16* __restrict__ Kg,
            const _Float16* __restrict__ Vg, const int* __restrict__ mask,
            _Float16* __restrict__ Og) {
  extern __shared__ char smem_raw[];
  _Float16* Qs  = (_Float16*)smem_raw;
  _Float16* Ksb[2];
  Ksb[0] = Qs + Q_HALVES;
  Ksb[1] = Ksb[0] + K_HALVES;
  _Float16* Ps  = Ksb[1] + K_HALVES;
  _Float16* Vs  = Ps + P_HALVES;
  float* mrow   = (float*)(Vs + V_HALVES); // [32] running row max (base-2)
  float* lrow   = mrow + 32;               // [32] running row sum
  float* cfac   = lrow + 32;               // [32] correction factor
  float* redmax = cfac + 32;               // [4][32]
  float* redsum = redmax + 128;            // [4][32]

  const int tid  = threadIdx.x;
  const int lane = tid & 31;
  const int wid  = tid >> 5;
  const int hi   = lane >> 4;
  const int nl   = lane & 15;
  const int bidx = blockIdx.y;
  const int q0   = blockIdx.x * 32;
  const size_t rowbase = (size_t)bidx * SEQ;

  const int smt = wid & 1;        // score m-tile (0/1)
  const int sg  = wid >> 1;       // score col group (0..3)
  const int sn0 = sg * 32;        // score col base within key block

  // Stage the 32x768 Q block once (8 threads/row, 96 halves = 12 b128 each).
  {
    const int r  = tid >> 3;
    const int c0 = (tid & 7) * 96;
    const _Float16* src = Qg + (rowbase + q0 + r) * DIM + c0;
#pragma unroll
    for (int t = 0; t < 12; ++t)
      copy16_g2l(Qs + r * QS_STR + c0 + 8 * t, src + 8 * t);
  }
  if (tid < 32) { mrow[tid] = -1e30f; lrow[tid] = 0.f; }

  v8f o[6][2];
#pragma unroll
  for (int nc = 0; nc < 6; ++nc)
#pragma unroll
    for (int mt = 0; mt < 2; ++mt) o[nc][mt] = vzero8();

  for (int kb = 0; kb < SEQ / 128; ++kb) {
    // ---- scores: S(32x128) = Q @ K^T, K double-buffered in 64-d chunks ----
    v8f sc[2];
    sc[0] = vzero8(); sc[1] = vzero8();

    stage_k(Ksb[0], Kg, rowbase, kb, 0, tid);
    async_join();
    __syncthreads();

#pragma unroll 2
    for (int dch = 0; dch < DIM / 64; ++dch) {
      const int cur = dch & 1;
      if (dch + 1 < DIM / 64)
        stage_k(Ksb[cur ^ 1], Kg, rowbase, kb, dch + 1, tid);
#pragma unroll
      for (int kk = 0; kk < 2; ++kk) {
        v16h a = load_frag_row(Qs + (16 * smt) * QS_STR + dch * 64 + kk * 32,
                               QS_STR, lane);
#pragma unroll
        for (int t = 0; t < 2; ++t) {
          v16h b = load_frag_row(Ksb[cur] + (sn0 + 16 * t) * KS_STR + kk * 32,
                                 KS_STR, lane);
          sc[t] = wmma_f16(a, b, sc[t]);
        }
      }
      async_join();
      __syncthreads();
    }

    // ---- scale (base-2), mask, block row-max ----
    float rm[8];
#pragma unroll
    for (int j = 0; j < 8; ++j) rm[j] = -1e30f;
#pragma unroll
    for (int t = 0; t < 2; ++t)
#pragma unroll
      for (int j = 0; j < 8; ++j) {
        float s = sc[t][j] * SM_SCALE_LOG2E;
        const int q    = q0 + 16 * smt + j + 8 * hi;
        const int kcol = kb * 128 + sn0 + 16 * t + nl;
        const int mk   = mask[((size_t)bidx * SEQ + q) * SEQ + kcol];
        s = (mk != 0) ? s : -1e30f;
        sc[t][j] = s;
        rm[j] = fmaxf(rm[j], s);
      }
#pragma unroll
    for (int j = 0; j < 8; ++j) {
      const float v = half_reduce_max(rm[j]);
      if (nl == 0) redmax[sg * 32 + 16 * smt + j + 8 * hi] = v;
    }
    __syncthreads();

    // ---- online softmax stats (one thread per query row) ----
    if (tid < 32) {
      const float bm = fmaxf(fmaxf(redmax[tid], redmax[32 + tid]),
                             fmaxf(redmax[64 + tid], redmax[96 + tid]));
      const float mo = mrow[tid];
      const float mn = fmaxf(mo, bm);
      const float cf = fast_exp2(mo - mn);
      cfac[tid] = cf;
      mrow[tid] = mn;
      lrow[tid] *= cf;
    }
    __syncthreads();

    // ---- P = exp2(s - m), stage to LDS, block row-sums ----
    float rs[8];
#pragma unroll
    for (int j = 0; j < 8; ++j) rs[j] = 0.f;
#pragma unroll
    for (int t = 0; t < 2; ++t)
#pragma unroll
      for (int j = 0; j < 8; ++j) {
        const int r = 16 * smt + j + 8 * hi;
        const float p = fast_exp2(sc[t][j] - mrow[r]);
        rs[j] += p;
        Ps[r * PS_STR + sn0 + 16 * t + nl] = (_Float16)p;
      }
#pragma unroll
    for (int j = 0; j < 8; ++j) {
      const float v = half_reduce_sum(rs[j]);
      if (nl == 0) redsum[sg * 32 + 16 * smt + j + 8 * hi] = v;
    }

    // rescale persistent O accumulators by the per-row correction
#pragma unroll
    for (int nc = 0; nc < 6; ++nc)
#pragma unroll
      for (int mt = 0; mt < 2; ++mt)
#pragma unroll
        for (int j = 0; j < 8; ++j)
          o[nc][mt][j] *= cfac[16 * mt + j + 8 * hi];
    __syncthreads();

    if (tid < 32)
      lrow[tid] += redsum[tid] + redsum[32 + tid] +
                   redsum[64 + tid] + redsum[96 + tid];

    // ---- O += P(32x128) @ V(128x768), V streamed in 128x128 tiles ----
    for (int nc = 0; nc < 6; ++nc) {
      const int vr  = tid >> 1;
      const int vc0 = (tid & 1) * 64;
      const _Float16* vsrc =
          Vg + (rowbase + kb * 128 + vr) * DIM + nc * 128 + vc0;
      __builtin_prefetch(vsrc + 128, 0, 1);  // next V chunk
#pragma unroll
      for (int t = 0; t < 8; ++t)
        copy16_g2l(Vs + vr * VS_STR + vc0 + 8 * t, vsrc + 8 * t);
      async_join();
      __syncthreads();
#pragma unroll
      for (int mt = 0; mt < 2; ++mt)
#pragma unroll
        for (int ks = 0; ks < 4; ++ks) {
          v16h a = load_frag_row(Ps + (16 * mt) * PS_STR + ks * 32, PS_STR,
                                 lane);
          v16h b = load_frag_b_kn(Vs + (ks * 32) * VS_STR + wid * 16, VS_STR,
                                  lane);
          o[nc][mt] = wmma_f16(a, b, o[nc][mt]);
        }
      __syncthreads();
    }
  }

  // ---- epilogue: normalize by row sums and store attn_out (f16) ----
#pragma unroll
  for (int nc = 0; nc < 6; ++nc)
#pragma unroll
    for (int mt = 0; mt < 2; ++mt)
#pragma unroll
      for (int j = 0; j < 8; ++j) {
        const int r = 16 * mt + j + 8 * hi;
        const float v = o[nc][mt][j] / lrow[r];
        Og[(rowbase + q0 + r) * DIM + nc * 128 + wid * 16 + nl] = (_Float16)v;
      }
}

// ---------------------------------------------------------------------------
// Kernel 4: out = x + attn_out @ Wo^T  (f32 output), 64x256 block tile.
// ---------------------------------------------------------------------------
__global__ void __launch_bounds__(256)
outproj_kernel(const _Float16* __restrict__ A, const float* __restrict__ W,
               const float* __restrict__ X, float* __restrict__ Out) {
  __shared__ _Float16 As[2][64][40];
  __shared__ _Float16 Bs[2][256][40];

  const int tid  = threadIdx.x;
  const int lane = tid & 31;
  const int wid  = tid >> 5;
  const int wm   = wid & 1;
  const int wn   = wid >> 1;
  const int m0   = blockIdx.x * 64;
  const int n0   = blockIdx.y * 256;

  v8f acc[2][4];
#pragma unroll
  for (int i = 0; i < 2; ++i)
#pragma unroll
    for (int j = 0; j < 4; ++j) acc[i][j] = vzero8();

  gemm_stage(As[0], Bs[0], A, W, m0, n0, 0, tid);
  async_join();
  __syncthreads();

#pragma unroll 2
  for (int it = 0; it < DIM / 32; ++it) {
    const int cur = it & 1;
    if (it + 1 < DIM / 32)
      gemm_stage(As[cur ^ 1], Bs[cur ^ 1], A, W, m0, n0, (it + 1) * 32, tid);

    v16h af[2], bf[4];
#pragma unroll
    for (int i = 0; i < 2; ++i)
      af[i] = load_frag_row(&As[cur][32 * wm + 16 * i][0], 40, lane);
#pragma unroll
    for (int j = 0; j < 4; ++j)
      bf[j] = load_frag_row(&Bs[cur][64 * wn + 16 * j][0], 40, lane);
#pragma unroll
    for (int i = 0; i < 2; ++i)
#pragma unroll
      for (int j = 0; j < 4; ++j) acc[i][j] = wmma_f16(af[i], bf[j], acc[i][j]);

    async_join();
    __syncthreads();
  }

  const int hi = lane >> 4, nl = lane & 15;
#pragma unroll
  for (int i = 0; i < 2; ++i)
#pragma unroll
    for (int j = 0; j < 4; ++j)
#pragma unroll
      for (int jj = 0; jj < 8; ++jj) {
        const size_t row = (size_t)(m0 + 32 * wm + 16 * i + jj + 8 * hi);
        const int col    = n0 + 64 * wn + 16 * j + nl;
        Out[row * DIM + col] = X[row * DIM + col] + acc[i][j][jj];
      }
}

// ---------------------------------------------------------------------------
// Host launcher.
// Inputs: 0:x 1:mask 2:Wq 3:Wk 4:Wv 5:Wo 6:ln_g 7:ln_b
// Workspace (f16): h | Q | K | V | attn_out, each 8192x768.
// ---------------------------------------------------------------------------
extern "C" void kernel_launch(void* const* d_in, const int* in_sizes, int n_in,
                              void* d_out, int out_size, void* d_ws,
                              size_t ws_size, hipStream_t stream) {
  const float* x    = (const float*)d_in[0];
  const int*   mask = (const int*)d_in[1];
  const float* Wq   = (const float*)d_in[2];
  const float* Wk   = (const float*)d_in[3];
  const float* Wv   = (const float*)d_in[4];
  const float* Wo   = (const float*)d_in[5];
  const float* ln_g = (const float*)d_in[6];
  const float* ln_b = (const float*)d_in[7];
  float* out = (float*)d_out;

  const size_t HN = (size_t)ROWS * DIM;
  _Float16* ws  = (_Float16*)d_ws;
  _Float16* h16 = ws;
  _Float16* q16 = ws + HN;
  _Float16* k16 = ws + 2 * HN;
  _Float16* v16 = ws + 3 * HN;
  _Float16* a16 = ws + 4 * HN;

  ln_cast_kernel<<<ROWS / 8, 256, 0, stream>>>(x, ln_g, ln_b, h16);

  const dim3 ggrid(ROWS / 64, DIM / 256);
  gemm_wt_f16_kernel<<<ggrid, 256, 0, stream>>>(h16, Wq, q16);
  gemm_wt_f16_kernel<<<ggrid, 256, 0, stream>>>(h16, Wk, k16);
  gemm_wt_f16_kernel<<<ggrid, 256, 0, stream>>>(h16, Wv, v16);

  attn_kernel<<<dim3(SEQ / 32, BATCH), 256, ATTN_SMEM_BYTES, stream>>>(
      q16, k16, v16, mask, a16);

  outproj_kernel<<<ggrid, 256, 0, stream>>>(a16, Wo, x, out);
}
